// SGLossOpt3_10084583211024
// MI455X (gfx1250) — matrix operand order repaired and checked by reading, not compiled, not run on hardware
//
#include <hip/hip_runtime.h>
#include <hip/hip_bf16.h>
#include <math.h>

// Problem constants (match reference setup_inputs)
#define BSZ   2048
#define LSZ   13
#define HSZ   768
#define NROWS (BSZ * LSZ)        // 26624 hidden rows == GEMM columns
#define INV_TEMP 20.0f           // 1 / 0.05
#define EPSN 1e-8f

// GEMM tiling
#define NTILES_TOTAL 1664        // 26624 / 16
#define NCHUNKS      64
#define NT_PER_CHUNK 26          // 64 * 26 = 1664
#define MGROUPS      16          // 128 M-tiles / 8 waves per block
#define TILE_BYTES   (16 * HSZ * 2)   // 24576 B: one 16-col B tile (contiguous in hn)

typedef __attribute__((ext_vector_type(16))) __bf16 v16bf;
typedef __attribute__((ext_vector_type(8)))  float  v8f;

__device__ __forceinline__ unsigned short f32_to_bf16(float f) {
    unsigned u = __float_as_uint(f);
    unsigned r = u + 0x7FFFu + ((u >> 16) & 1u);   // round-to-nearest-even
    return (unsigned short)(r >> 16);
}

// ---------------- zero accumulator ----------------
__global__ void zero_kernel(float* __restrict__ p, int n) {
    int i = blockIdx.x * blockDim.x + threadIdx.x;
    if (i < n) p[i] = 0.0f;
}

// ---------------- row normalize: fp32 -> bf16, store 1/max(||x||,eps) ----------------
__global__ void __launch_bounds__(256) nrm_kernel(const float* __restrict__ src,
                                                  unsigned short* __restrict__ dst,
                                                  float* __restrict__ inv_out,
                                                  int nrows) {
    int wid  = (blockIdx.x * blockDim.x + threadIdx.x) >> 5;
    int lane = threadIdx.x & 31;
    if (wid >= nrows) return;                 // uniform per wave
    const float* p = src + (size_t)wid * HSZ;
    float x[24];
    float ss = 0.0f;
#pragma unroll
    for (int j = 0; j < 24; ++j) { x[j] = p[lane + 32 * j]; ss += x[j] * x[j]; }
#pragma unroll
    for (int m = 16; m >= 1; m >>= 1) ss += __shfl_xor(ss, m, 32);
    float inv = 1.0f / fmaxf(sqrtf(ss), EPSN);
    unsigned short* q = dst + (size_t)wid * HSZ;
#pragma unroll
    for (int j = 0; j < 24; ++j) q[lane + 32 * j] = f32_to_bf16(x[j] * inv);
    if (lane == 0) inv_out[wid] = inv;
}

// ---------------- async copy of one contiguous 24KB B tile into LDS ----------------
// 256 threads x 6 x b128 = 24576 bytes. hn rows are K-contiguous, so a 16-col
// B tile is one contiguous block at byte offset n0*1536.
__device__ __forceinline__ void prefetch_tile(const unsigned short* __restrict__ hn,
                                              int n0, unsigned lds_off, int tid) {
    unsigned goff = (unsigned)n0 * (unsigned)(HSZ * 2) + (unsigned)tid * 16u;
    unsigned loff = lds_off + (unsigned)tid * 16u;
#pragma unroll
    for (int i = 0; i < 6; ++i) {
        asm volatile("global_load_async_to_lds_b128 %0, %1, %2"
                     :: "v"(loff + (unsigned)(i * 4096)),
                        "v"(goff + (unsigned)(i * 4096)),
                        "s"(hn)
                     : "memory");
    }
}

// ---------------- big GEMM + exp + mask + masked row-sum ----------------
// Block = 8 waves = 8 M-tiles sharing each B tile via LDS (double-buffered,
// filled by global_load_async_to_lds_b128). A stripe pinned in registers
// (24 x v16bf = 192 VGPRs). 24 x v_wmma_f32_16x16x32_bf16 per tile.
__global__ void __launch_bounds__(256) gemm_kernel(const unsigned short* __restrict__ An,  // [2048,768] bf16
                                                   const unsigned short* __restrict__ Bn,  // [26624,768] bf16
                                                   float* __restrict__ s_acc) {            // [2048]
    __shared__ __align__(16) unsigned short smem[2 * 16 * HSZ];   // 2 x 24KB

    int tid    = threadIdx.x;
    int wave   = tid >> 5;
    int lane   = tid & 31;
    int mgroup = blockIdx.x & (MGROUPS - 1);
    int chunk  = blockIdx.x >> 4;                 // 0..63
    int m0     = (mgroup * 8 + wave) * 16;
    int half   = lane >> 4;
    int l15    = lane & 15;
    int rbase  = m0 + half * 8;

    // A fragment base: 16x32 bf16 layout — lanes 0-15: K {0..7,16..23}; lanes 16-31: K {8..15,24..31}
    const unsigned short* arow = An + (size_t)(m0 + l15) * HSZ + half * 8;

    unsigned lds_base = (unsigned)(size_t)(&smem[0]);

    int ntile0 = chunk * NT_PER_CHUNK;
    prefetch_tile(Bn, ntile0 * 16, lds_base, tid);   // overlap with A register fill

    // Pin the whole 16x768 A stripe in registers: 24 fragments of 16 bf16.
    v16bf afrag[24];
#pragma unroll
    for (int kk = 0; kk < 24; ++kk) {
        union { uint4 u[2]; v16bf v; } ua;
        ua.u[0] = *(const uint4*)(arow + kk * 32);
        ua.u[1] = *(const uint4*)(arow + kk * 32 + 16);
        afrag[kk] = ua.v;
    }

    float acc[8];
#pragma unroll
    for (int r = 0; r < 8; ++r) acc[r] = 0.0f;

    for (int nt = 0; nt < NT_PER_CHUNK; ++nt) {
        asm volatile("s_wait_asynccnt 0" ::: "memory");  // my slice of tile nt is in LDS
        __syncthreads();                                  // whole tile nt ready; everyone past nt-1
        if (nt + 1 < NT_PER_CHUNK)
            prefetch_tile(Bn, (ntile0 + nt + 1) * 16,
                          lds_base + (unsigned)(((nt + 1) & 1) * TILE_BYTES), tid);

        // B fragment from LDS: lane n (0-15): col n, K {0..7,16..23}; lane n+16: col n, K {8..15,24..31}
        const unsigned short* bcol = smem + (size_t)((nt & 1) * 16 * HSZ)
                                          + (size_t)l15 * HSZ + half * 8;

        v8f c = {0.f, 0.f, 0.f, 0.f, 0.f, 0.f, 0.f, 0.f};
#pragma unroll
        for (int kk = 0; kk < 24; ++kk) {
            union { uint4 u[2]; v16bf v; } ub;
            ub.u[0] = *(const uint4*)(bcol + kk * 32);       // ds_load_b128 x2
            ub.u[1] = *(const uint4*)(bcol + kk * 32 + 16);
            c = __builtin_amdgcn_wmma_f32_16x16x32_bf16(
                    false, afrag[kk], false, ub.v, (short)0, c, false, false);
        }

        int ncol = (ntile0 + nt) * 16 + l15;             // this lane's global column
#pragma unroll
        for (int r = 0; r < 8; ++r) {
            float v = __expf(c[r] * INV_TEMP);
            if ((ncol & (BSZ - 1)) == rbase + r) v = 0.0f;   // mask: c % B == i
            acc[r] += v;
        }
    }

    // reduce across the 16 lanes of each half (columns), then atomically add per row
#pragma unroll
    for (int m = 1; m < 16; m <<= 1) {
#pragma unroll
        for (int r = 0; r < 8; ++r) acc[r] += __shfl_xor(acc[r], m, 16);
    }
    if (l15 == 0) {
#pragma unroll
        for (int r = 0; r < 8; ++r) unsafeAtomicAdd(&s_acc[rbase + r], acc[r]);
    }
}

// ---------------- diagonal sims from raw fp32 inputs (full precision) ----------------
__global__ void __launch_bounds__(256) diag_kernel(const float* __restrict__ cls,
                                                   const float* __restrict__ hid,
                                                   const float* __restrict__ inv_c,
                                                   const float* __restrict__ inv_h,
                                                   float* __restrict__ sim) {
    int wid  = blockIdx.x * 8 + (threadIdx.x >> 5);   // 0..26623 exactly
    int lane = threadIdx.x & 31;
    int i = wid / LSZ;
    const float* c = cls + (size_t)i * HSZ;
    const float* h = hid + (size_t)wid * HSZ;
    float ssum = 0.0f;
#pragma unroll
    for (int j = 0; j < 24; ++j) ssum += c[lane + 32 * j] * h[lane + 32 * j];
#pragma unroll
    for (int m = 16; m >= 1; m >>= 1) ssum += __shfl_xor(ssum, m, 32);
    if (lane == 0) sim[wid] = __expf(ssum * inv_c[i] * inv_h[wid] * INV_TEMP);
}

// ---------------- final reduction: mean log1p(s_i / sim_il) ----------------
__global__ void loss_kernel(const float* __restrict__ s_acc,
                            const float* __restrict__ sim,
                            float* __restrict__ out) {
    __shared__ float red[1024];
    int tid = threadIdx.x;
    float sum = 0.0f;
    for (int idx = tid; idx < NROWS; idx += 1024) {
        int i = idx / LSZ;
        sum += log1pf(s_acc[i] / sim[idx]);
    }
    red[tid] = sum;
    __syncthreads();
    for (int o = 512; o > 0; o >>= 1) {
        if (tid < o) red[tid] += red[tid + o];
        __syncthreads();
    }
    if (tid == 0) out[0] = red[0] * (1.0f / (float)NROWS);
}

extern "C" void kernel_launch(void* const* d_in, const int* in_sizes, int n_in,
                              void* d_out, int out_size, void* d_ws, size_t ws_size,
                              hipStream_t stream) {
    const float* cls = (const float*)d_in[0];   // [2048, 768]
    const float* hid = (const float*)d_in[1];   // [2048, 13, 768] == [26624, 768]
    float* out = (float*)d_out;

    // workspace layout (all 16B aligned), total ~44.3 MB
    char* ws = (char*)d_ws;
    unsigned short* cn    = (unsigned short*)(ws);                 //  3,145,728 B
    unsigned short* hn    = (unsigned short*)(ws + 3145728);       // 40,894,464 B
    float*          inv_c = (float*)(ws + 44040192);               //      8,192 B
    float*          inv_h = (float*)(ws + 44048384);               //    106,496 B
    float*          sim   = (float*)(ws + 44154880);               //    106,496 B
    float*          s_acc = (float*)(ws + 44261376);               //      8,192 B

    zero_kernel<<<2, 1024, 0, stream>>>(s_acc, BSZ);
    nrm_kernel<<<BSZ / 8, 256, 0, stream>>>(cls, cn, inv_c, BSZ);
    nrm_kernel<<<NROWS / 8, 256, 0, stream>>>(hid, hn, inv_h, NROWS);
    gemm_kernel<<<MGROUPS * NCHUNKS, 256, 0, stream>>>(cn, hn, s_acc);
    diag_kernel<<<NROWS / 8, 256, 0, stream>>>(cls, hid, inv_c, inv_h, sim);
    loss_kernel<<<1, 1024, 0, stream>>>(s_acc, sim, out);
}
